// Net_74208444940775
// MI455X (gfx1250) — matrix operand order repaired and checked by reading, not compiled, not run on hardware
//
#include <hip/hip_runtime.h>

#ifndef __has_builtin
#define __has_builtin(x) 0
#endif

typedef __bf16 bf16_t;
typedef __attribute__((ext_vector_type(16))) __bf16 v16bf;
typedef __attribute__((ext_vector_type(8)))  float  v8f;
typedef int v4i __attribute__((vector_size(16)));
typedef __attribute__((address_space(1))) v4i* as1_v4i_p;
typedef __attribute__((address_space(3))) v4i* as3_v4i_p;

#define DD 128
#define GN_EPS 1e-5f

// ---------------- fragment loaders (wave32, v_wmma_f32_16x16x32_bf16) ----------------
// A tile 16x32: lane half h holds row m=lane&15, K = 8h..8h+7 (elems 0..7) and
// 16+8h..16+8h+7 (elems 8..15), relative to chunk base k0.  (ISA 7.12.2)
__device__ __forceinline__ v16bf frag_from_f32(const float* row0, int ld, int k0) {
  int lane = threadIdx.x & 31;
  int m = lane & 15, kh = (lane >> 4) << 3;
  const float* r = row0 + (size_t)m * ld + k0 + kh;
  v16bf f;
#pragma unroll
  for (int i = 0; i < 8; ++i) {
    f[i]     = (bf16_t)r[i];
    f[i + 8] = (bf16_t)r[i + 16];
  }
  return f;
}

__device__ __forceinline__ v16bf frag_from_bf16(const bf16_t* row0, int ld, int k0) {
  int lane = threadIdx.x & 31;
  int m = lane & 15, kh = (lane >> 4) << 3;
  const bf16_t* r = row0 + (size_t)m * ld + k0 + kh;
  v16bf f;
#pragma unroll
  for (int i = 0; i < 8; ++i) { f[i] = r[i]; f[i + 8] = r[i + 16]; }
  return f;
}

// B tile 32x16 for y = x @ W^T : column n = W row n; lane half h holds K=8h..8h+7 (elems 0..7) and 16+8h..16+8h+7 (elems 8..15).
__device__ __forceinline__ v16bf fragB_from_bf16(const bf16_t* W, int ldw, int ncol0, int k0) {
  int lane = threadIdx.x & 31;
  int n = lane & 15, h = lane >> 4;
  const bf16_t* r = W + (size_t)(ncol0 + n) * ldw + k0 + h * 8;
  v16bf f;
#pragma unroll
  for (int i = 0; i < 8; ++i) { f[i] = r[i]; f[i + 8] = r[i + 16]; }
  return f;
}

// ---------------- 16x128 GEMM tile: D = A(16x128) @ W^T, W row-major (128x128) ----------------
__device__ __forceinline__ void gemm_stage(const v16bf A[4], const bf16_t* W, int ldw, int koff,
                                           float acc[8][8]) {
#pragma unroll
  for (int nt = 0; nt < 8; ++nt) {
    v8f c = {0.f, 0.f, 0.f, 0.f, 0.f, 0.f, 0.f, 0.f};
#pragma unroll
    for (int kc = 0; kc < 4; ++kc) {
      v16bf b = fragB_from_bf16(W, ldw, nt * 16, koff + kc * 32);
      c = __builtin_amdgcn_wmma_f32_16x16x32_bf16(false, A[kc], false, b, (short)0, c,
                                                  false, false);
    }
#pragma unroll
    for (int v = 0; v < 8; ++v) acc[nt][v] = c[v];
  }
}

// ---------------- fused GroupNorm(+opt ReLU), rows reduced across 16-lane half-waves ----------------
__device__ __forceinline__ void gn_relu(float acc[8][8], const float* g, const float* b,
                                        bool do_relu) {
  int lane = threadIdx.x & 31;
  int n0 = lane & 15;
  float gv[8], bv[8];
#pragma unroll
  for (int nt = 0; nt < 8; ++nt) { gv[nt] = g[nt * 16 + n0]; bv[nt] = b[nt * 16 + n0]; }
#pragma unroll
  for (int v = 0; v < 8; ++v) {
    float s = 0.f, s2 = 0.f;
#pragma unroll
    for (int nt = 0; nt < 8; ++nt) { float x = acc[nt][v]; s += x; s2 += x * x; }
#pragma unroll
    for (int msk = 1; msk < 16; msk <<= 1) {
      s  += __shfl_xor(s,  msk, 32);
      s2 += __shfl_xor(s2, msk, 32);
    }
    float mean = s * (1.f / 128.f);
    float var  = s2 * (1.f / 128.f) - mean * mean;
    float inv  = rsqrtf(var + GN_EPS);
#pragma unroll
    for (int nt = 0; nt < 8; ++nt) {
      float y = (acc[nt][v] - mean) * inv * gv[nt] + bv[nt];
      acc[nt][v] = do_relu ? fmaxf(y, 0.f) : y;
    }
  }
}

__device__ __forceinline__ void store_lds(bf16_t* la, const float acc[8][8]) {
  int lane = threadIdx.x & 31; int n0 = lane & 15; int h = lane >> 4;
#pragma unroll
  for (int nt = 0; nt < 8; ++nt)
#pragma unroll
    for (int v = 0; v < 8; ++v)
      la[(h * 8 + v) * DD + nt * 16 + n0] = (bf16_t)acc[nt][v];
}

__device__ __forceinline__ void store_global(float* p, const float acc[8][8]) {
  int lane = threadIdx.x & 31; int n0 = lane & 15; int h = lane >> 4;
#pragma unroll
  for (int nt = 0; nt < 8; ++nt)
#pragma unroll
    for (int v = 0; v < 8; ++v)
      p[(size_t)(h * 8 + v) * DD + nt * 16 + n0] = acc[nt][v];
}

// ---------------- block-cooperative weight staging: global -> LDS ----------------
// Uses CDNA5 async-to-LDS (ASYNCcnt) when the toolchain exposes it; else plain copy.
__device__ __forceinline__ void stage_weight_lds(bf16_t* ldst, const bf16_t* gsrc, int bytes) {
  int t = threadIdx.x;  // blockDim.x == 128
#if __has_builtin(__builtin_amdgcn_global_load_async_to_lds_b128)
  for (int off = t * 16; off < bytes; off += 128 * 16) {
    const char* g = (const char*)gsrc + off;
    char* l = (char*)ldst + off;
    __builtin_amdgcn_global_load_async_to_lds_b128((as1_v4i_p)g, (as3_v4i_p)l, 0, 0);
  }
#if __has_builtin(__builtin_amdgcn_s_wait_asynccnt)
  __builtin_amdgcn_s_wait_asynccnt(0);
#else
  asm volatile("s_wait_asynccnt 0" ::: "memory");
#endif
#else
  for (int off = t * 16; off < bytes; off += 128 * 16)
    *(uint4*)((char*)ldst + off) = *(const uint4*)((const char*)gsrc + off);
#endif
  __syncthreads();
}

// ---------------- utility kernels ----------------
__global__ void k_cvt(const float* __restrict__ src, bf16_t* __restrict__ dst, int n) {
  int i = blockIdx.x * blockDim.x + threadIdx.x;
  if (i < n) dst[i] = (bf16_t)src[i];
}
// split ctx_w1 (128x384) into three contiguous 128x128 bf16 blocks
__global__ void k_pack3(const float* __restrict__ src, bf16_t* __restrict__ d0,
                        bf16_t* __restrict__ d1, bf16_t* __restrict__ d2) {
  int i = blockIdx.x * blockDim.x + threadIdx.x;
  if (i >= 128 * 384) return;
  int n = i / 384, k = i % 384;
  bf16_t v = (bf16_t)src[i];
  if (k < 128)      d0[n * DD + k]       = v;
  else if (k < 256) d1[n * DD + k - 128] = v;
  else              d2[n * DD + k - 256] = v;
}
__global__ void k_zero(float* p, int n) {
  int i = blockIdx.x * blockDim.x + threadIdx.x;
  if (i < n) p[i] = 0.f;
}

// ---------------- per-node pass 1: q_part, ctx_part, agts @ agt_w^T ----------------
__global__ void __launch_bounds__(128)
k_node1(const bf16_t* __restrict__ agts_bf, const bf16_t* __restrict__ ctx_bf,
        const bf16_t* __restrict__ wq, const bf16_t* __restrict__ cw1q,
        const bf16_t* __restrict__ cw1c, const bf16_t* __restrict__ aw,
        const float* __restrict__ qg, const float* __restrict__ qb,
        float* __restrict__ q_part, float* __restrict__ c_part, float* __restrict__ agtsW,
        int ntiles) {
  __shared__ __align__(16) bf16_t ldsA[4][16 * DD];
  int wv = threadIdx.x >> 5;
  int tile = blockIdx.x * 4 + wv;
  if (tile >= ntiles) return;
  int base = tile * 16;
  bf16_t* la = ldsA[wv];
  float acc[8][8];
  v16bf A[4], A2[4];

#pragma unroll
  for (int kc = 0; kc < 4; ++kc) A[kc] = frag_from_bf16(agts_bf + (size_t)base * DD, DD, kc * 32);

  // qn = relu(gn(agts @ q_w^T))
  gemm_stage(A, wq, DD, 0, acc);
  gn_relu(acc, qg, qb, true);
  store_lds(la, acc);

  // q_part = qn @ ctx_w1[:,128:256]^T
#pragma unroll
  for (int kc = 0; kc < 4; ++kc) A2[kc] = frag_from_bf16(la, DD, kc * 32);
  gemm_stage(A2, cw1q, DD, 0, acc);
  store_global(q_part + (size_t)base * DD, acc);

  // agtsW = agts @ agt_w^T  (A frags still live)
  gemm_stage(A, aw, DD, 0, acc);
  store_global(agtsW + (size_t)base * DD, acc);

  // ctx_part = ctx @ ctx_w1[:,256:384]^T
#pragma unroll
  for (int kc = 0; kc < 4; ++kc) A2[kc] = frag_from_bf16(ctx_bf + (size_t)base * DD, DD, kc * 32);
  gemm_stage(A2, cw1c, DD, 0, acc);
  store_global(c_part + (size_t)base * DD, acc);
}

// ---------------- per-edge pass: dist MLP -> ctx_w1 d-block + biases -> GN -> scatter ----------------
__global__ void __launch_bounds__(128)
k_edge(const int* __restrict__ hi, const int* __restrict__ wi,
       const float* __restrict__ actr, const float* __restrict__ cctr,
       const float* __restrict__ w1, const float* __restrict__ b1,
       const bf16_t* __restrict__ dw2, const bf16_t* __restrict__ cw1d,
       const float* __restrict__ dg, const float* __restrict__ dbt,
       const float* __restrict__ cg, const float* __restrict__ cbt,
       const float* __restrict__ q_part, const float* __restrict__ c_part,
       float* __restrict__ accum, int E, int ntiles) {
  __shared__ __align__(16) bf16_t ldsW[DD * DD];      // dist_w2 staged once per block (32KB)
  __shared__ __align__(16) bf16_t ldsA[4][16 * DD];   // per-wave re-layout tile (16KB)

  // prefetch stage-2 weights into cache while staging stage-1 weights into LDS
  __builtin_prefetch((const char*)cw1d + threadIdx.x * 256, 0, 3);
  stage_weight_lds(ldsW, dw2, DD * DD * 2);           // async-to-LDS + s_wait_asynccnt + barrier

  int wv = threadIdx.x >> 5;
  int tile = blockIdx.x * 4 + wv;
  if (tile >= ntiles) return;
  int base = tile * 16;
  int lane = threadIdx.x & 31, n0 = lane & 15, h = lane >> 4;
  bf16_t* la = ldsA[wv];

  // per-lane row (both halves carry row m = lane&15 of the A tile)
  int e = base + n0;
  float dx = 0.f, dy = 0.f;
  if (e < E) {
    int hv = hi[e], wvv = wi[e];
    dx = actr[2 * hv]     - cctr[2 * wvv];
    dy = actr[2 * hv + 1] - cctr[2 * wvv + 1];
  }

  // h1 = relu([dx,dy] @ dist_w1^T + b1), built directly as A fragments
  v16bf A[4];
#pragma unroll
  for (int kc = 0; kc < 4; ++kc) {
#pragma unroll
    for (int i = 0; i < 8; ++i) {
      int K1 = kc * 32 + h * 8 + i;
      int K2 = K1 + 16;
      float u1 = fmaxf(dx * w1[2 * K1] + dy * w1[2 * K1 + 1] + b1[K1], 0.f);
      float u2 = fmaxf(dx * w1[2 * K2] + dy * w1[2 * K2 + 1] + b1[K2], 0.f);
      A[kc][i] = (bf16_t)u1;
      A[kc][i + 8] = (bf16_t)u2;
    }
  }

  float acc[8][8];
  // h2 = relu(gn(h1 @ dist_w2^T)), B-fragments served from LDS
  gemm_stage(A, ldsW, DD, 0, acc);
  gn_relu(acc, dg, dbt, true);
  store_lds(la, acc);

  // row indices for this lane's D-layout rows (rows 8h..8h+7)
  int hrow[8], wrow[8]; bool rv[8];
#pragma unroll
  for (int v = 0; v < 8; ++v) {
    int e3 = base + h * 8 + v;
    rv[v] = e3 < E;
    hrow[v] = rv[v] ? hi[e3] : 0;
    wrow[v] = rv[v] ? wi[e3] : 0;
  }

  // c = relu(gn(h2 @ ctx_w1[:,0:128]^T + q_part[hi] + ctx_part[wi]))
  v16bf A2[4];
#pragma unroll
  for (int kc = 0; kc < 4; ++kc) A2[kc] = frag_from_bf16(la, DD, kc * 32);
  gemm_stage(A2, cw1d, DD, 0, acc);
#pragma unroll
  for (int v = 0; v < 8; ++v) {
    if (rv[v]) {
#pragma unroll
      for (int nt = 0; nt < 8; ++nt)
        acc[nt][v] += q_part[(size_t)hrow[v] * DD + nt * 16 + n0] +
                      c_part[(size_t)wrow[v] * DD + nt * 16 + n0];
    }
  }
  gn_relu(acc, cg, cbt, true);

  // scatter-add into per-node accumulator (ctx_w2 applied per-node later)
#pragma unroll
  for (int v = 0; v < 8; ++v) {
    if (rv[v]) {
#pragma unroll
      for (int nt = 0; nt < 8; ++nt)
        atomicAdd(&accum[(size_t)hrow[v] * DD + nt * 16 + n0], acc[nt][v]);
    }
  }
}

// ---------------- per-node pass 2: ctx_w2, norm, lin, residual ----------------
__global__ void __launch_bounds__(128)
k_node2(const float* __restrict__ accum, const float* __restrict__ agtsW,
        const float* __restrict__ agts,
        const bf16_t* __restrict__ cw2, const bf16_t* __restrict__ lw,
        const float* __restrict__ ng, const float* __restrict__ nb,
        const float* __restrict__ lg, const float* __restrict__ lbt,
        float* __restrict__ out, int ntiles) {
  __shared__ __align__(16) bf16_t ldsA[4][16 * DD];
  int wv = threadIdx.x >> 5;
  int tile = blockIdx.x * 4 + wv;
  if (tile >= ntiles) return;
  int base = tile * 16;
  int lane = threadIdx.x & 31, n0 = lane & 15, h = lane >> 4;
  bf16_t* la = ldsA[wv];
  float acc[8][8];
  v16bf A[4];

  // a = agtsW + accum_c @ ctx_w2^T
#pragma unroll
  for (int kc = 0; kc < 4; ++kc) A[kc] = frag_from_f32(accum + (size_t)base * DD, DD, kc * 32);
  gemm_stage(A, cw2, DD, 0, acc);
#pragma unroll
  for (int nt = 0; nt < 8; ++nt)
#pragma unroll
    for (int v = 0; v < 8; ++v)
      acc[nt][v] += agtsW[(size_t)(base + h * 8 + v) * DD + nt * 16 + n0];

  // a = relu(gn(a))
  gn_relu(acc, ng, nb, true);
  store_lds(la, acc);

  // y = gn(a @ lin_w^T); out = relu(y + agts)
#pragma unroll
  for (int kc = 0; kc < 4; ++kc) A[kc] = frag_from_bf16(la, DD, kc * 32);
  gemm_stage(A, lw, DD, 0, acc);
  gn_relu(acc, lg, lbt, false);
#pragma unroll
  for (int nt = 0; nt < 8; ++nt)
#pragma unroll
    for (int v = 0; v < 8; ++v) {
      size_t idx = (size_t)(base + h * 8 + v) * DD + nt * 16 + n0;
      out[idx] = fmaxf(acc[nt][v] + agts[idx], 0.f);
    }
}

extern "C" void kernel_launch(void* const* d_in, const int* in_sizes, int n_in,
                              void* d_out, int out_size, void* d_ws, size_t ws_size,
                              hipStream_t stream) {
  const float* agts = (const float*)d_in[0];
  const float* ctx  = (const float*)d_in[1];
  const float* actr = (const float*)d_in[2];
  const float* cctr = (const float*)d_in[3];
  const float* dw1  = (const float*)d_in[4];
  const float* db1  = (const float*)d_in[5];
  const float* dw2  = (const float*)d_in[6];
  const float* dg   = (const float*)d_in[7];
  const float* dbt  = (const float*)d_in[8];
  const float* qw   = (const float*)d_in[9];
  const float* qg   = (const float*)d_in[10];
  const float* qb   = (const float*)d_in[11];
  const float* cw1  = (const float*)d_in[12];
  const float* cg   = (const float*)d_in[13];
  const float* cbt  = (const float*)d_in[14];
  const float* cw2  = (const float*)d_in[15];
  const float* aw   = (const float*)d_in[16];
  const float* ng   = (const float*)d_in[17];
  const float* nb   = (const float*)d_in[18];
  const float* lw   = (const float*)d_in[19];
  const float* lg   = (const float*)d_in[20];
  const float* lbt  = (const float*)d_in[21];
  const int*   hi   = (const int*)d_in[22];
  const int*   wi   = (const int*)d_in[23];
  int E = in_sizes[22];
  int N = in_sizes[0] / DD;  // 20000 (multiple of 16)
  float* out = (float*)d_out;

  // ---- workspace layout ----
  // [0, 256KB): 8 contiguous 128x128 bf16 weight blocks
  bf16_t* wsb   = (bf16_t*)d_ws;
  bf16_t* b_qw   = wsb + 0 * 16384;
  bf16_t* b_dw2  = wsb + 1 * 16384;
  bf16_t* b_cw1d = wsb + 2 * 16384;  // ctx_w1[:,  0:128]
  bf16_t* b_cw1q = wsb + 3 * 16384;  // ctx_w1[:,128:256]
  bf16_t* b_cw1c = wsb + 4 * 16384;  // ctx_w1[:,256:384]
  bf16_t* b_cw2  = wsb + 5 * 16384;
  bf16_t* b_aw   = wsb + 6 * 16384;
  bf16_t* b_lw   = wsb + 7 * 16384;
  // bf16 copies of node features
  size_t SZ = (size_t)N * DD;
  bf16_t* agts_bf = wsb + 8 * 16384;
  bf16_t* ctx_bf  = agts_bf + SZ;
  // f32 intermediates
  float* f32ws = (float*)(ctx_bf + SZ);
  float* q_part = f32ws;
  float* c_part = f32ws + SZ;
  float* agtsW  = f32ws + 2 * SZ;
  float* accum  = f32ws + 3 * SZ;

  k_cvt<<<(16384 + 255) / 256, 256, 0, stream>>>(qw,  b_qw,  16384);
  k_cvt<<<(16384 + 255) / 256, 256, 0, stream>>>(dw2, b_dw2, 16384);
  k_pack3<<<(49152 + 255) / 256, 256, 0, stream>>>(cw1, b_cw1d, b_cw1q, b_cw1c);
  k_cvt<<<(16384 + 255) / 256, 256, 0, stream>>>(cw2, b_cw2, 16384);
  k_cvt<<<(16384 + 255) / 256, 256, 0, stream>>>(aw,  b_aw,  16384);
  k_cvt<<<(16384 + 255) / 256, 256, 0, stream>>>(lw,  b_lw,  16384);
  k_cvt<<<((int)SZ + 255) / 256, 256, 0, stream>>>(agts, agts_bf, (int)SZ);
  k_cvt<<<((int)SZ + 255) / 256, 256, 0, stream>>>(ctx,  ctx_bf,  (int)SZ);
  k_zero<<<((int)SZ + 255) / 256, 256, 0, stream>>>(accum, (int)SZ);

  int ntN = N / 16;
  int nbN = (ntN + 3) / 4;
  k_node1<<<nbN, 128, 0, stream>>>(agts_bf, ctx_bf, b_qw, b_cw1q, b_cw1c, b_aw, qg, qb,
                                   q_part, c_part, agtsW, ntN);
  int ntE = (E + 15) / 16;
  int nbE = (ntE + 3) / 4;
  if (nbE > 0)
    k_edge<<<nbE, 128, 0, stream>>>(hi, wi, actr, cctr, dw1, db1, b_dw2, b_cw1d,
                                    dg, dbt, cg, cbt, q_part, c_part, accum, E, ntE);
  k_node2<<<nbN, 128, 0, stream>>>(accum, agtsW, agts, b_cw2, b_lw,
                                   ng, nb, lg, lbt, out, ntN);
}
